// AttentionLayer_6725918785800
// MI455X (gfx1250) — matrix operand order repaired and checked by reading, not compiled, not run on hardware
//
#include <hip/hip_runtime.h>
#include <cmath>

// ---------------------------------------------------------------------------
// Attention: O = softmax(Q K^T) V with Q/K/V = X @ W{q,k,v} + b (no 1/sqrt(d))
// B=4, L=4096, D=768, fp32 in/out.  Compute path: bf16 WMMA, fp32 accumulate.
// ---------------------------------------------------------------------------

#define Bc 4
#define Lc 4096
#define Dc 768

typedef __attribute__((ext_vector_type(16))) __bf16    v16bf;
typedef __attribute__((ext_vector_type(8)))  float     v8f;
typedef __attribute__((ext_vector_type(4)))  float     f32x4;
typedef __attribute__((ext_vector_type(4)))  unsigned  u32x4;
typedef __attribute__((ext_vector_type(2)))  unsigned  u32x2;

union Frag {                  // 32 bytes = one WMMA 16-bit A or B operand
    v16bf bf;
    u32x4 q[2];
};

__device__ __forceinline__ unsigned short f2bf(float f) {
    union { float f; unsigned u; } x; x.f = f;
    unsigned r = x.u + 0x7FFFu + ((x.u >> 16) & 1u);   // round-to-nearest-even
    return (unsigned short)(r >> 16);
}

__device__ __forceinline__ unsigned pack2(float a, float b) {
    return (unsigned)f2bf(a) | ((unsigned)f2bf(b) << 16);
}

__device__ __forceinline__ v8f wmma_bf16(const Frag& a, const Frag& b, v8f c) {
    return __builtin_amdgcn_wmma_f32_16x16x32_bf16(
        /*neg_a=*/false, a.bf, /*neg_b=*/false, b.bf,
        /*c_mod=*/(short)0, c, /*reuse_a=*/false, /*reuse_b=*/false);
}

// ---------------------------------------------------------------------------
// Kernel 1: fused QKV projection.  C[16384x768] = X @ W (+bias), x3 weights.
// Q,K stored row-major bf16; V stored TRANSPOSED per batch (Vt[D][L]) so the
// flash kernel can build PV B-fragments with contiguous global reads.
// grid = (128, 6, 3)   block = 256 (8 waves)
// ---------------------------------------------------------------------------
__global__ __launch_bounds__(256)
void qkv_proj_kernel(const float* __restrict__ X,
                     const float* __restrict__ Wq, const float* __restrict__ bq,
                     const float* __restrict__ Wk, const float* __restrict__ bk,
                     const float* __restrict__ Wv, const float* __restrict__ bv,
                     unsigned short* __restrict__ outQ,
                     unsigned short* __restrict__ outK,
                     unsigned short* __restrict__ outVt) {
    __shared__ __align__(16) unsigned short Xl[128][32];   // bf16 X tile
    __shared__ __align__(16) unsigned short Wt[128][32];   // bf16 W^T tile

    const int mat  = blockIdx.z;
    const float* __restrict__ W  = (mat == 0) ? Wq : (mat == 1) ? Wk : Wv;
    const float* __restrict__ bi = (mat == 0) ? bq : (mat == 1) ? bk : bv;

    const int row0 = blockIdx.x * 128;      // over B*L = 16384
    const int col0 = blockIdx.y * 128;      // over D   = 768
    const int tid  = threadIdx.x;
    const int wave = tid >> 5, lane = tid & 31;
    const int lm = lane & 15, lh = lane >> 4;
    const int mb0 = (wave >> 1) * 32;       // wave's 32 output rows
    const int nb0 = (wave & 1) * 64;        // wave's 64 output cols

    v8f acc[2][4];
    #pragma unroll
    for (int m = 0; m < 2; ++m)
        #pragma unroll
        for (int n = 0; n < 4; ++n)
            #pragma unroll
            for (int i = 0; i < 8; ++i) acc[m][n][i] = 0.f;

    for (int kk = 0; kk < Dc; kk += 32) {
        __syncthreads();
        // stage X tile (128x32 f32 -> bf16): b128 loads, 8B packed LDS stores
        #pragma unroll
        for (int i = tid * 4; i < 128 * 32; i += 256 * 4) {
            int r = i >> 5, c = i & 31;
            const float* xp = X + (size_t)(row0 + r) * Dc + kk + c;
            f32x4 v = *(const f32x4*)xp;
            if (kk + 32 < Dc)
                __builtin_prefetch(xp + 32, 0, 1);         // next k-chunk
            u32x2 p; p.x = pack2(v.x, v.y); p.y = pack2(v.z, v.w);
            *(u32x2*)&Xl[r][c] = p;
        }
        // stage W chunk transposed (32x128 -> Wt[n][k]): b128 loads + scatter
        #pragma unroll
        for (int i = tid * 4; i < 32 * 128; i += 256 * 4) {
            int k = i >> 7, n = i & 127;
            const float* wp = W + (size_t)(kk + k) * Dc + col0 + n;
            f32x4 v = *(const f32x4*)wp;
            if (kk + 32 < Dc)
                __builtin_prefetch(wp + 32 * Dc, 0, 1);    // next k-chunk
            Wt[n + 0][k] = f2bf(v.x);
            Wt[n + 1][k] = f2bf(v.y);
            Wt[n + 2][k] = f2bf(v.z);
            Wt[n + 3][k] = f2bf(v.w);
        }
        __syncthreads();

        #pragma unroll
        for (int mb = 0; mb < 2; ++mb) {
            Frag a;
            const unsigned short* ap = &Xl[mb0 + mb * 16 + lm][0];
            a.q[0] = *(const u32x4*)(ap + lh * 8);
            a.q[1] = *(const u32x4*)(ap + 16 + lh * 8);
            #pragma unroll
            for (int nb = 0; nb < 4; ++nb) {
                Frag bfr;
                const unsigned short* bp = &Wt[nb0 + nb * 16 + lm][0];
                bfr.q[0] = *(const u32x4*)(bp + lh * 8);
                bfr.q[1] = *(const u32x4*)(bp + 16 + lh * 8);
                acc[mb][nb] = wmma_bf16(a, bfr, acc[mb][nb]);
            }
        }
    }

    // epilogue: bias add, convert, store (V goes out transposed per batch)
    #pragma unroll
    for (int mb = 0; mb < 2; ++mb)
        #pragma unroll
        for (int nb = 0; nb < 4; ++nb) {
            int col = col0 + nb0 + nb * 16 + lm;
            float bv_ = bi[col];
            #pragma unroll
            for (int i = 0; i < 8; ++i) {
                int row = row0 + mb0 + mb * 16 + i + 8 * lh;    // global B*L row
                unsigned short h = f2bf(acc[mb][nb][i] + bv_);
                if (mat == 0) {
                    outQ[(size_t)row * Dc + col] = h;
                } else if (mat == 1) {
                    outK[(size_t)row * Dc + col] = h;
                } else {
                    int b = row >> 12, l = row & (Lc - 1);
                    outVt[(size_t)b * Dc * Lc + (size_t)col * Lc + l] = h;
                }
            }
        }
}

// ---------------------------------------------------------------------------
// Kernel 2: flash attention.  One WG = 16 query rows of one batch.
// 8 waves: wave w owns D-slice [w*96, w*96+96) of both the QK^T reduction
// (partials summed into LDS via ds_add_f32) and the O accumulator.
// K-fragments and V-fragments come straight from global (L2-resident, 6 MB
// each per batch on a 192 MB L2); next tile is prefetched while this tile's
// WMMA + softmax runs.  grid = (256, 4)  block = 256.
// ---------------------------------------------------------------------------
__global__ __launch_bounds__(256)
void flash_attn_kernel(const unsigned short* __restrict__ Q,
                       const unsigned short* __restrict__ K,
                       const unsigned short* __restrict__ Vt,
                       float* __restrict__ O) {
    __shared__ __align__(16) float          S_lds[16][64];       // shared scores
    __shared__ __align__(16) unsigned short P_lds[8][16][64];    // per-wave P
    __shared__ __align__(16) float          stat_lds[8][16];     // per-wave bcast

    const int q0   = blockIdx.x * 16;
    const int b    = blockIdx.y;
    const int tid  = threadIdx.x;
    const int wave = tid >> 5, lane = tid & 31;
    const int lm = lane & 15, lh = lane >> 4;
    const int d0 = wave * 96;                       // this wave's D slice

    const unsigned short* __restrict__ Qb  = Q  + (size_t)b * Lc * Dc;
    const unsigned short* __restrict__ Kb  = K  + (size_t)b * Lc * Dc;
    const unsigned short* __restrict__ Vtb = Vt + (size_t)b * Dc * Lc;

    // Q A-fragments for this wave's k-slice: hoisted out of the tile loop
    Frag qa[3];
    {
        const unsigned short* qrow = Qb + (size_t)(q0 + lm) * Dc;
        #pragma unroll
        for (int c = 0; c < 3; ++c) {
            int kc = d0 + c * 32;
            qa[c].q[0] = *(const u32x4*)(qrow + kc + lh * 8);
            qa[c].q[1] = *(const u32x4*)(qrow + kc + 16 + lh * 8);
        }
    }

    v8f o[6];
    #pragma unroll
    for (int n = 0; n < 6; ++n)
        #pragma unroll
        for (int i = 0; i < 8; ++i) o[n][i] = 0.f;

    float m_run = -3.0e38f, l_run = 0.f;            // lanes 0..15: row stats

    for (int kt = 0; kt < Lc; kt += 64) {
        // overlap: pull next tile's K/V lines toward L2/L0 while we compute
        if (kt + 64 < Lc) {
            __builtin_prefetch(Kb + (size_t)(kt + 64 + (lane << 1)) * Dc + d0, 0, 1);
            __builtin_prefetch(Vtb + (size_t)(d0 + (lane & 15)) * Lc + kt + 64, 0, 1);
        }

        __syncthreads();                            // S_lds free from last tile
        for (int i = tid; i < 16 * 64; i += 256) ((float*)S_lds)[i] = 0.f;
        __syncthreads();

        // ---- partial S = Q[:, d0:d0+96] @ K[:, d0:d0+96]^T  (12 WMMAs) ----
        v8f s[4];
        #pragma unroll
        for (int n = 0; n < 4; ++n)
            #pragma unroll
            for (int i = 0; i < 8; ++i) s[n][i] = 0.f;

        #pragma unroll
        for (int c = 0; c < 3; ++c) {
            int kc = d0 + c * 32;
            #pragma unroll
            for (int nb = 0; nb < 4; ++nb) {
                Frag kf;
                const unsigned short* kp =
                    Kb + (size_t)(kt + nb * 16 + lm) * Dc + kc;
                kf.q[0] = *(const u32x4*)(kp + lh * 8);
                kf.q[1] = *(const u32x4*)(kp + 16 + lh * 8);
                s[nb] = wmma_bf16(qa[c], kf, s[nb]);
            }
        }
        // combine partials across the 8 waves
        #pragma unroll
        for (int nb = 0; nb < 4; ++nb)
            #pragma unroll
            for (int i = 0; i < 8; ++i)
                atomicAdd(&S_lds[i + 8 * lh][nb * 16 + lm], s[nb][i]);
        __syncthreads();

        // ---- online softmax (redundant per wave; lanes 0..15 = rows) ----
        if (lane < 16) {
            int r = lane;
            float tmax = -3.0e38f;
            #pragma unroll 8
            for (int j = 0; j < 64; ++j) tmax = fmaxf(tmax, S_lds[r][j]);
            float m_new = fmaxf(m_run, tmax);
            float alpha = __expf(m_run - m_new);
            float ssum  = 0.f;
            #pragma unroll 8
            for (int j = 0; j < 64; j += 2) {
                float p0 = __expf(S_lds[r][j]     - m_new);
                float p1 = __expf(S_lds[r][j + 1] - m_new);
                ssum += p0 + p1;
                *(unsigned*)&P_lds[wave][r][j] = (unsigned)f2bf(p0) |
                                                 ((unsigned)f2bf(p1) << 16);
            }
            l_run = alpha * l_run + ssum;
            m_run = m_new;
            stat_lds[wave][r] = alpha;
        }
        __builtin_amdgcn_wave_barrier();            // DS in-order within wave

        // rescale O accumulators (row M = i + 8*lh)
        float af[8];
        #pragma unroll
        for (int i = 0; i < 8; ++i) af[i] = stat_lds[wave][i + 8 * lh];
        #pragma unroll
        for (int nb = 0; nb < 6; ++nb)
            #pragma unroll
            for (int i = 0; i < 8; ++i) o[nb][i] *= af[i];

        // ---- O += P @ V[kt:kt+64, d0:d0+96]   (12 WMMAs, V^T from L2) ----
        #pragma unroll
        for (int c = 0; c < 2; ++c) {
            Frag pa;
            const unsigned short* pr = &P_lds[wave][lm][0];
            pa.q[0] = *(const u32x4*)(pr + c * 32 + lh * 8);
            pa.q[1] = *(const u32x4*)(pr + c * 32 + 16 + lh * 8);
            #pragma unroll
            for (int nb = 0; nb < 6; ++nb) {
                Frag vf;
                const unsigned short* vp =
                    Vtb + (size_t)(d0 + nb * 16 + lm) * Lc + kt + c * 32;
                vf.q[0] = *(const u32x4*)(vp + lh * 8);
                vf.q[1] = *(const u32x4*)(vp + 16 + lh * 8);
                o[nb] = wmma_bf16(pa, vf, o[nb]);
            }
        }
    }

    // ---- normalize by softmax denominator and store fp32 output ----
    if (lane < 16) stat_lds[wave][lane] = l_run;
    __builtin_amdgcn_wave_barrier();
    float inv[8];
    #pragma unroll
    for (int i = 0; i < 8; ++i) inv[i] = 1.0f / stat_lds[wave][i + 8 * lh];

    #pragma unroll
    for (int nb = 0; nb < 6; ++nb)
        #pragma unroll
        for (int i = 0; i < 8; ++i) {
            int row = q0 + i + 8 * lh;
            int col = d0 + nb * 16 + lm;
            O[((size_t)b * Lc + row) * Dc + col] = o[nb][i] * inv[i];
        }
}

// ---------------------------------------------------------------------------
extern "C" void kernel_launch(void* const* d_in, const int* in_sizes, int n_in,
                              void* d_out, int out_size, void* d_ws, size_t ws_size,
                              hipStream_t stream) {
    const float* X  = (const float*)d_in[0];
    const float* Wq = (const float*)d_in[1];
    const float* bq = (const float*)d_in[2];
    const float* Wk = (const float*)d_in[3];
    const float* bk = (const float*)d_in[4];
    const float* Wv = (const float*)d_in[5];
    const float* bv = (const float*)d_in[6];

    const size_t elems = (size_t)Bc * Lc * Dc;      // 12.58M per tensor
    unsigned short* wsQ  = (unsigned short*)d_ws;
    unsigned short* wsK  = wsQ + elems;
    unsigned short* wsVt = wsK + elems;             // total 75.5 MB bf16

    qkv_proj_kernel<<<dim3(Bc * Lc / 128, Dc / 128, 3), 256, 0, stream>>>(
        X, Wq, bq, Wk, bk, Wv, bv, wsQ, wsK, wsVt);

    flash_attn_kernel<<<dim3(Lc / 16, Bc), 256, 0, stream>>>(
        wsQ, wsK, wsVt, (float*)d_out);
}
